// CrossCrossSelfAtt_5549097746879
// MI455X (gfx1250) — matrix operand order repaired and checked by reading, not compiled, not run on hardware
//
#include <hip/hip_runtime.h>
#include <hip/hip_bf16.h>

// ---------------------------------------------------------------------------
// Types for CDNA5 WMMA (wave32, 16x16x32 bf16 -> f32)
// ---------------------------------------------------------------------------
typedef __bf16 bf16_t;
typedef __attribute__((ext_vector_type(16))) __bf16 v16bf;
typedef __attribute__((ext_vector_type(8)))  float  v8f;

// Single packed convert: dst[15:0] = bf16(lo), dst[31:16] = bf16(hi).
// (The union/cast version lowered to a 6-op merge chain; force the ISA op.)
__device__ __forceinline__ unsigned pack_bf16x2(float lo, float hi) {
  unsigned r;
  asm("v_cvt_pk_bf16_f32 %0, %1, %2" : "=v"(r) : "v"(lo), "v"(hi));
  return r;
}

// ---------------------------------------------------------------------------
// GEMM: C[M,N] = A[M,768] @ W[768,N] + bias[N]   (f32 in/out, bf16 WMMA core)
// Workgroup: 256 threads (8 waves). Tile: 128(M) x 64(N), BK = 32.
// Waves arranged 4(M) x 2(N); each wave owns a 32x32 block = 2x2 WMMA tiles.
// LDS double-buffered: global loads for tile k+1 are issued between the
// barrier and the WMMAs of tile k, so HBM/L2 latency overlaps the matrix ops.
// A staged row-major bf16; W staged TRANSPOSED (Bsm[n][k]) so both fragment
// reads are two contiguous 16B runs (k=8h.., k=16+8h..) -> ds_load_b128 pairs.
// ---------------------------------------------------------------------------
#define GEMM_K 768
#define LDA 40   // padded LDS stride in bf16 elems (80B, keeps 16B alignment)
#define LDB 40

__global__ __launch_bounds__(256)
void gemm_bias_kernel(const float* __restrict__ A, const float* __restrict__ W,
                      const float* __restrict__ bias, float* __restrict__ C,
                      int M, int N) {
  __shared__ __align__(16) bf16_t Asm[2][128 * LDA];
  __shared__ __align__(16) bf16_t Bsm[2][64 * LDB];   // transposed: [n][k]

  const int tid   = threadIdx.x;
  const int mBase = blockIdx.x * 128;
  const int nBase = blockIdx.y * 64;
  const int wave  = tid >> 5;
  const int lane  = tid & 31;
  const int wm    = wave & 3;      // 0..3  (M direction)
  const int wn    = wave >> 2;     // 0..1  (N direction)
  const int hlf   = lane >> 4;     // 0/1   (lane half)
  const int r     = lane & 15;

  // --- staging assignments (fixed per thread) ---
  // A: 4 float4 chunks; chunk c = tid + 256*i -> row = c>>3, kq = (c&7)*4
  // B: one column n = tid%64, k-octet kc = tid/64 (0..3): 8 strided loads
  const int bn = tid & 63;
  const int bk = (tid >> 6) << 3;   // 0,8,16,24

  float4 ar[4];
  float  br[8];

  auto loadTiles = [&](int k0) {
#pragma unroll
    for (int i = 0; i < 4; ++i) {
      const int c   = tid + 256 * i;
      const int row = c >> 3;
      const int kq  = (c & 7) << 2;
      ar[i] = *(const float4*)(A + (size_t)(mBase + row) * GEMM_K + k0 + kq);
    }
    const float* wp = W + (size_t)(k0 + bk) * N + nBase + bn;
#pragma unroll
    for (int j = 0; j < 8; ++j) br[j] = wp[(size_t)j * N];
  };

  auto storeTiles = [&](int buf) {
#pragma unroll
    for (int i = 0; i < 4; ++i) {
      const int c   = tid + 256 * i;
      const int row = c >> 3;
      const int kq  = (c & 7) << 2;
      uint2 pv;
      pv.x = pack_bf16x2(ar[i].x, ar[i].y);
      pv.y = pack_bf16x2(ar[i].z, ar[i].w);
      *(uint2*)&Asm[buf][row * LDA + kq] = pv;           // ds_store_b64
    }
    uint4 bv;
    bv.x = pack_bf16x2(br[0], br[1]);
    bv.y = pack_bf16x2(br[2], br[3]);
    bv.z = pack_bf16x2(br[4], br[5]);
    bv.w = pack_bf16x2(br[6], br[7]);
    *(uint4*)&Bsm[buf][bn * LDB + bk] = bv;              // ds_store_b128
  };

  v8f acc[2][2] = {};

  loadTiles(0);
  int buf = 0;
  for (int k0 = 0; k0 < GEMM_K; k0 += 32) {
    storeTiles(buf);            // waits on loadcnt here, hidden by prior WMMAs
    __syncthreads();

    if (k0 + 32 < GEMM_K) loadTiles(k0 + 32);   // overlap with WMMAs below

    // ---- build fragments (ISA 7.12.2 layouts) ----
    // A (16x32): lane holds row M=lane%16; elems 0..7 -> K=8h..8h+7,
    //            elems 8..15 -> K=16+8h..16+8h+7.
    v16bf afrag[2], bfrag[2];
#pragma unroll
    for (int mi = 0; mi < 2; ++mi) {
      const int row = wm * 32 + mi * 16 + r;
      union { v16bf v; uint4 u[2]; } t;
      t.u[0] = *(const uint4*)&Asm[buf][row * LDA + 8 * hlf];
      t.u[1] = *(const uint4*)&Asm[buf][row * LDA + 16 + 8 * hlf];
      afrag[mi] = t.v;
    }
    // B (32x16): lane holds col N=lane%16; elems 0..7 -> K=8h..8h+7,
    //            elems 8..15 -> K=16+8h..16+8h+7 (Bsm is stored transposed).
#pragma unroll
    for (int ni = 0; ni < 2; ++ni) {
      const int col = wn * 32 + ni * 16 + r;
      union { v16bf v; uint4 u[2]; } t;
      t.u[0] = *(const uint4*)&Bsm[buf][col * LDB + 8 * hlf];
      t.u[1] = *(const uint4*)&Bsm[buf][col * LDB + 16 + 8 * hlf];
      bfrag[ni] = t.v;
    }

#pragma unroll
    for (int mi = 0; mi < 2; ++mi)
#pragma unroll
      for (int ni = 0; ni < 2; ++ni)
        acc[mi][ni] = __builtin_amdgcn_wmma_f32_16x16x32_bf16(
            false, afrag[mi], false, bfrag[ni], (short)0, acc[mi][ni], false, false);

    buf ^= 1;
  }

  // ---- epilogue: C/D layout -> VGPR j holds M = 8*hlf + j, N = lane%16 ----
#pragma unroll
  for (int mi = 0; mi < 2; ++mi)
#pragma unroll
    for (int ni = 0; ni < 2; ++ni) {
      const int col = nBase + wn * 32 + ni * 16 + r;
      const float bv = bias ? bias[col] : 0.f;
#pragma unroll
      for (int j = 0; j < 8; ++j) {
        const int row = mBase + wm * 32 + mi * 16 + hlf * 8 + j;
        C[(size_t)row * N + col] = acc[mi][ni][j] + bv;
      }
    }
}

// ---------------------------------------------------------------------------
// Row LayerNorm over D=768: y = (x - mu) * rsqrt(var + eps) * g + be
// One workgroup (256 threads) per row; 3 elements per thread.
// ---------------------------------------------------------------------------
__global__ __launch_bounds__(256)
void ln_kernel(const float* __restrict__ x, const float* __restrict__ g,
               const float* __restrict__ be, float* __restrict__ y) {
  const size_t row = blockIdx.x;
  const int t = threadIdx.x;
  __shared__ float red[256];

  const float v0 = x[row * 768 + t];
  const float v1 = x[row * 768 + t + 256];
  const float v2 = x[row * 768 + t + 512];

  red[t] = v0 + v1 + v2;
  __syncthreads();
  for (int o = 128; o > 0; o >>= 1) { if (t < o) red[t] += red[t + o]; __syncthreads(); }
  const float mu = red[0] * (1.f / 768.f);
  __syncthreads();

  const float d0 = v0 - mu, d1 = v1 - mu, d2 = v2 - mu;
  red[t] = d0 * d0 + d1 * d1 + d2 * d2;
  __syncthreads();
  for (int o = 128; o > 0; o >>= 1) { if (t < o) red[t] += red[t + o]; __syncthreads(); }
  const float rstd = rsqrtf(red[0] * (1.f / 768.f) + 1e-6f);

  y[row * 768 + t]       = d0 * rstd * g[t]       + be[t];
  y[row * 768 + t + 256] = d1 * rstd * g[t + 256] + be[t + 256];
  y[row * 768 + t + 512] = d2 * rstd * g[t + 512] + be[t + 512];
}

// ---------------------------------------------------------------------------
// Small-width row softmax (w = 12 for cross-attn, 4 for self-attn)
// ---------------------------------------------------------------------------
__global__ void softmax_kernel(float* __restrict__ x, int rows, int w) {
  const int rIdx = blockIdx.x * blockDim.x + threadIdx.x;
  if (rIdx >= rows) return;
  float* p = x + (size_t)rIdx * w;
  float mx = p[0];
  for (int i = 1; i < w; ++i) mx = fmaxf(mx, p[i]);
  float s = 0.f;
  for (int i = 0; i < w; ++i) { const float e = __expf(p[i] - mx); p[i] = e; s += e; }
  const float inv = 1.f / s;
  for (int i = 0; i < w; ++i) p[i] *= inv;
}

// ---------------------------------------------------------------------------
// MSDA sampling: out[b,q,h,d] = sum_{l,p} aw * bilinear(val_l, loc)
// One workgroup per (b,q): 768 threads = (head h = t/48, channel d = t%48).
// Offsets/weights staged once in LDS. NLEV = 3 (levels 64/32/16) or 1 (32x32).
// ---------------------------------------------------------------------------
template <int NLEV>
__global__ __launch_bounds__(768)
void msda_sample_kernel(const float* __restrict__ val, const float* __restrict__ off,
                        const float* __restrict__ aw, float* __restrict__ out,
                        int srcLen) {
  constexpr int P = 4;
  constexpr int offW = 16 * NLEV * P * 2;
  constexpr int awW  = 16 * NLEV * P;
  const int bq = blockIdx.x;          // b*1024 + q
  const int t  = threadIdx.x;         // h*48 + d
  const int h  = t / 48;
  const int d  = t % 48;
  const int q  = bq & 1023;
  const int b  = bq >> 10;

  __shared__ float soff[offW];
  __shared__ float saw[awW];
  if (t < offW) soff[t] = off[(size_t)bq * offW + t];
  if (t < awW)  saw[t]  = aw[(size_t)bq * awW + t];
  __syncthreads();

  const float refx = ((q & 31) + 0.5f) * (1.f / 32.f);
  const float refy = ((q >> 5) + 0.5f) * (1.f / 32.f);

  float acc = 0.f;
#pragma unroll
  for (int l = 0; l < NLEV; ++l) {
    const int H  = (NLEV == 1) ? 32 : (l == 0 ? 64 : (l == 1 ? 32 : 16));
    const int Wd = H;
    const int st = (NLEV == 1) ? 0 : (l == 0 ? 0 : (l == 1 ? 4096 : 5120));
    const float* vb = val + ((size_t)b * srcLen + st) * 768 + h * 48 + d;
#pragma unroll
    for (int p = 0; p < P; ++p) {
      const float ox  = soff[((h * NLEV + l) * P + p) * 2 + 0];
      const float oy  = soff[((h * NLEV + l) * P + p) * 2 + 1];
      const float wgt = saw[(h * NLEV + l) * P + p];
      const float px = (refx + ox / (float)Wd) * (float)Wd - 0.5f;
      const float py = (refy + oy / (float)H)  * (float)H  - 0.5f;
      const float x0f = floorf(px), y0f = floorf(py);
      const float wx = px - x0f, wy = py - y0f;
      const int x0 = (int)x0f, y0 = (int)y0f;

      auto tap = [&](int xi, int yi) -> float {
        if (xi < 0 || xi >= Wd || yi < 0 || yi >= H) return 0.f;
        return vb[(size_t)(yi * Wd + xi) * 768];
      };
      const float v00 = tap(x0,     y0);
      const float v10 = tap(x0 + 1, y0);
      const float v01 = tap(x0,     y0 + 1);
      const float v11 = tap(x0 + 1, y0 + 1);
      acc += wgt * (v00 * (1.f - wx) * (1.f - wy) + v10 * wx * (1.f - wy) +
                    v01 * (1.f - wx) * wy          + v11 * wx * wy);
    }
  }
  out[(size_t)bq * 768 + t] = acc;
}

// ---------------------------------------------------------------------------
// out[i] (+)= x[i] * g[i % 768]
// ---------------------------------------------------------------------------
__global__ void scale_add_kernel(float* __restrict__ out, const float* __restrict__ x,
                                 const float* __restrict__ g, int n, int accumulate) {
  const int i = blockIdx.x * blockDim.x + threadIdx.x;
  if (i >= n) return;
  const float v = x[i] * g[i % 768];
  out[i] = accumulate ? (out[i] + v) : v;
}

// ---------------------------------------------------------------------------
// Host orchestration
// ---------------------------------------------------------------------------
namespace {
constexpr int D_   = 768;
constexpr int B_   = 8;
constexpr int LQ   = 1024;          // 32*32 queries
constexpr int LIN  = 5376;          // 64^2 + 32^2 + 16^2
constexpr int MQ   = B_ * LQ;       // 8192
constexpr int MF   = B_ * LIN;      // 43008
constexpr int NHH  = 16;

struct MsdaP { const float *Wa, *Wo, *Woff, *Wv, *ba, *bo, *boff, *bv; };
struct LnP   { const float *W, *b, *be, *g; };
}  // namespace

extern "C" void kernel_launch(void* const* d_in, const int* in_sizes, int n_in,
                              void* d_out, int out_size, void* d_ws, size_t ws_size,
                              hipStream_t stream) {
  (void)in_sizes; (void)n_in; (void)out_size; (void)ws_size;

  const float* query  = (const float*)d_in[0];
  const float* feat   = (const float*)d_in[1];
  const float* feat_1 = (const float*)d_in[2];
  // d_in[3]=spatial_shapes, d_in[4]=level_start_index (constants, hardcoded)

  // params flattened as a jax pytree: dict keys in alphabetical order.
  // top: ca, ca1, fn, fn1, g1, g1_1, norm, norm1, qn, qn1, sa, sa1
  // msda dict: Wa, Wo, Woff, Wv, ba, bo, boff, bv
  // linnorm dict: W, b, be, g           norm dict: b, g
  int ix = 5;
  auto getM = [&](MsdaP& m) {
    m.Wa = (const float*)d_in[ix++]; m.Wo   = (const float*)d_in[ix++];
    m.Woff = (const float*)d_in[ix++]; m.Wv = (const float*)d_in[ix++];
    m.ba = (const float*)d_in[ix++]; m.bo   = (const float*)d_in[ix++];
    m.boff = (const float*)d_in[ix++]; m.bv = (const float*)d_in[ix++];
  };
  auto getL = [&](LnP& l) {
    l.W = (const float*)d_in[ix++]; l.b  = (const float*)d_in[ix++];
    l.be = (const float*)d_in[ix++]; l.g = (const float*)d_in[ix++];
  };
  MsdaP ca, ca1, sa, sa1; LnP fn, fn1, qn, qn1;
  const float *g1, *g1_1, *norm_b, *norm_g, *norm1_b, *norm1_g;
  getM(ca); getM(ca1); getL(fn); getL(fn1);
  g1   = (const float*)d_in[ix++];
  g1_1 = (const float*)d_in[ix++];
  norm_b  = (const float*)d_in[ix++]; norm_g  = (const float*)d_in[ix++];
  norm1_b = (const float*)d_in[ix++]; norm1_g = (const float*)d_in[ix++];
  getL(qn); getL(qn1); getM(sa); getM(sa1);

  // ---- workspace layout (f32) ----
  float* ws = (float*)d_ws;
  size_t o = 0;
  float* bufQ   = ws + o; o += (size_t)MQ * D_;        // qn / qn1 output
  float* bufF   = ws + o; o += (size_t)MF * D_;        // fn / fn1 output
  float* bufV   = ws + o; o += (size_t)MF * D_;        // cross-attn values
  float* bufV2  = ws + o; o += (size_t)MQ * D_;        // self-attn values
  float* bufOff = ws + o; o += (size_t)MQ * 384;       // sampling offsets
  float* bufAw  = ws + o; o += (size_t)MQ * 192;       // attention weights
  float* bufS   = ws + o; o += (size_t)MQ * D_;        // sampled output (pre-Wo)
  float* bufA   = ws + o; o += (size_t)MQ * D_;        // attn scratch (reused x4)
  float* bufN1  = ws + o; o += (size_t)MQ * D_;        // attn1 (live for sa1!)
  float* bufN2  = ws + o; o += (size_t)MQ * D_;        // attn1_1

  float* out = (float*)d_out;
  const int nOut = MQ * D_;

  auto gemm = [&](const float* A, const float* W, const float* bias, float* C,
                  int M, int N) {
    dim3 grid(M / 128, N / 64);
    gemm_bias_kernel<<<grid, 256, 0, stream>>>(A, W, bias, C, M, N);
  };
  auto ln = [&](const float* x, const float* g, const float* be, float* y, int rows) {
    ln_kernel<<<rows, 256, 0, stream>>>(x, g, be, y);
  };
  auto smax = [&](float* x, int w) {
    const int rows = MQ * NHH;
    softmax_kernel<<<(rows + 255) / 256, 256, 0, stream>>>(x, rows, w);
  };

  // ===== block 1: cross-attn (ca) =====
  gemm(query, qn.W, qn.b, bufQ, MQ, D_);  ln(bufQ, qn.g, qn.be, bufQ, MQ);
  gemm(feat,  fn.W, fn.b, bufF, MF, D_);  ln(bufF, fn.g, fn.be, bufF, MF);
  gemm(bufF, ca.Wv,   ca.bv,   bufV,   MF, D_);
  gemm(bufQ, ca.Woff, ca.boff, bufOff, MQ, 384);
  gemm(bufQ, ca.Wa,   ca.ba,   bufAw,  MQ, 192);  smax(bufAw, 12);
  msda_sample_kernel<3><<<MQ, 768, 0, stream>>>(bufV, bufOff, bufAw, bufS, LIN);
  gemm(bufS, ca.Wo, ca.bo, bufA, MQ, D_);                    // attn
  ln(bufA, norm_g, norm_b, bufN1, MQ);                       // attn1
  scale_add_kernel<<<(nOut + 255) / 256, 256, 0, stream>>>(out, bufA, g1, nOut, 0);

  // ===== block 2: self-attn (sa): q = src = attn1 =====
  gemm(bufN1, sa.Wv,   sa.bv,   bufV2,  MQ, D_);
  gemm(bufN1, sa.Woff, sa.boff, bufOff, MQ, 128);
  gemm(bufN1, sa.Wa,   sa.ba,   bufAw,  MQ, 64);  smax(bufAw, 4);
  msda_sample_kernel<1><<<MQ, 768, 0, stream>>>(bufV2, bufOff, bufAw, bufS, LQ);
  gemm(bufS, sa.Wo, sa.bo, bufA, MQ, D_);                    // attn2
  scale_add_kernel<<<(nOut + 255) / 256, 256, 0, stream>>>(out, bufA, g1, nOut, 1);

  // ===== block 3: cross-attn (ca1): q = LN1(attn2), src = LN(feat_1) =====
  gemm(bufA,   qn1.W, qn1.b, bufQ, MQ, D_);  ln(bufQ, qn1.g, qn1.be, bufQ, MQ);
  gemm(feat_1, fn1.W, fn1.b, bufF, MF, D_);  ln(bufF, fn1.g, fn1.be, bufF, MF);
  gemm(bufF, ca1.Wv,   ca1.bv,   bufV,   MF, D_);
  gemm(bufQ, ca1.Woff, ca1.boff, bufOff, MQ, 384);
  gemm(bufQ, ca1.Wa,   ca1.ba,   bufAw,  MQ, 192);  smax(bufAw, 12);
  msda_sample_kernel<3><<<MQ, 768, 0, stream>>>(bufV, bufOff, bufAw, bufS, LIN);
  gemm(bufS, ca1.Wo, ca1.bo, bufA, MQ, D_);                  // attn_1
  ln(bufA, norm1_g, norm1_b, bufN2, MQ);                     // attn1_1
  scale_add_kernel<<<(nOut + 255) / 256, 256, 0, stream>>>(out, bufA, g1_1, nOut, 1);

  // ===== block 4: self-attn (sa1): q = attn1_1, src = attn1 (per reference!) =====
  gemm(bufN1, sa1.Wv,   sa1.bv,   bufV2,  MQ, D_);
  gemm(bufN2, sa1.Woff, sa1.boff, bufOff, MQ, 128);
  gemm(bufN2, sa1.Wa,   sa1.ba,   bufAw,  MQ, 64);  smax(bufAw, 4);
  msda_sample_kernel<1><<<MQ, 768, 0, stream>>>(bufV2, bufOff, bufAw, bufS, LQ);
  gemm(bufS, sa1.Wo, sa1.bo, bufA, MQ, D_);                  // attn2_1
  scale_add_kernel<<<(nOut + 255) / 256, 256, 0, stream>>>(out, bufA, g1_1, nOut, 1);
}
